// VarianceAdaptor_34153579938107
// MI455X (gfx1250) — compile-verified, matrix operand order checked by reading, not run on hardware
//
#include <hip/hip_runtime.h>
#include <hip/hip_bf16.h>

// ---------------------------------------------------------------------------
// VarianceAdaptor (FastSpeech2) for MI455X / gfx1250
// Convs as bf16 WMMA GEMMs (v_wmma_f32_16x16x32_bf16, fp32 accum) with a
// software-pipelined k-loop; fused ReLU+LayerNorm; NT-store expansion.
// ---------------------------------------------------------------------------

typedef __attribute__((ext_vector_type(16))) __bf16 v16bf;
typedef __attribute__((ext_vector_type(8)))  __bf16 v8bf;
typedef __attribute__((ext_vector_type(8)))  float  v8f;
typedef __attribute__((ext_vector_type(4)))  float  v4f;

#define B_      32
#define T_      512
#define H_      384
#define F_      384
#define KD_     1152      // 3 * 384 (im2col K-dim)
#define MAXLEN_ 4096
#define LDA_    392       // 384 + 8 bf16 pad -> conflict-free ds_load_b128

// ---------------------------------------------------------------------------
// Weight prep: fp32 (KD_ x F_) -> bf16 transposed (F_ x KD_) so each lane's
// B-fragment is two contiguous 16B global loads.
// ---------------------------------------------------------------------------
__global__ __launch_bounds__(256)
void k_wt(const float* __restrict__ w, __bf16* __restrict__ wt)
{
    int i = blockIdx.x * 256 + threadIdx.x;
    if (i >= KD_ * F_) return;
    int kg = i / F_;
    int f  = i - kg * F_;
    wt[(size_t)f * KD_ + kg] = (__bf16)w[i];
}

// ---------------------------------------------------------------------------
// Fused conv1d(K=3, pad=1) + bias + ReLU + LayerNorm.
// One workgroup: 16 tokens x 384 features. 8 waves, 3 16x16 N-tiles each.
// A staged in LDS (bf16, zero-padded halo). B read from N-major bf16 weights.
// k-loop is 2-stage software pipelined: step k+1 fragments are issued before
// step k's WMMAs so the pre-WMMA s_wait covers latency-hidden loads.
// ---------------------------------------------------------------------------
template<bool ABF>
__global__ __launch_bounds__(256)
void k_conv_ln(const void* __restrict__ Ain,            // fp32 or bf16 [B,T,384]
               const __bf16* __restrict__ Wt,           // [F_=384][KD_=1152] bf16
               const float* __restrict__ bias,
               const float* __restrict__ gamma,
               const float* __restrict__ beta,
               __bf16* __restrict__ Out)                // [B,T,384] bf16
{
    __shared__ __attribute__((aligned(16))) __bf16 ldsA[18 * LDA_];
    __shared__ float ldsC[16 * F_];

    const int tid  = threadIdx.x;
    const int lane = tid & 31;
    const int wave = tid >> 5;
    const int b    = blockIdx.x >> 5;        // T_/16 = 32 tiles per batch
    const int t0   = (blockIdx.x & 31) << 4;

    // Stage 18 rows (t0-1 .. t0+16) x 384 cols into LDS as bf16, zero halo.
    for (int i = tid; i < 18 * H_; i += 256) {
        int r = i / H_, c = i - r * H_;
        int t = t0 + r - 1;
        float v = 0.f;
        if (t >= 0 && t < T_) {
            size_t g = ((size_t)b * T_ + t) * H_ + c;
            v = ABF ? (float)((const __bf16*)Ain)[g] : ((const float*)Ain)[g];
        }
        ldsA[r * LDA_ + c] = (__bf16)v;
    }
    __syncthreads();

    const int hl   = lane >> 4;       // wave half (K-split in fragment layout)
    const int m    = lane & 15;       // row for A, col for B
    const int ncol = lane & 15;
    const int n0   = wave * 48;       // 3 N-tiles per wave

    // A fragment: elems 0..7 -> K = kb + hl*8 + {0..7}; 8..15 -> +16.
    auto lda_frag = [&](int kk) {
        const int kb  = kk * 32;
        const int tap = kb / H_;              // 384 = 12*32: chunk stays in a tap
        const int c0  = kb - tap * H_ + hl * 8;
        const __bf16* ap = &ldsA[(m + tap) * LDA_ + c0];
        union { v16bf v; v8bf h[2]; } u;
        u.h[0] = *(const v8bf*)(ap);
        u.h[1] = *(const v8bf*)(ap + 16);
        return u.v;
    };
    // B fragment from N-major weights: two contiguous b128 loads per lane.
    auto ldb_frag = [&](int kk, int col) {
        const __bf16* bp = Wt + (size_t)col * KD_ + kk * 32 + hl * 8;
        union { v16bf v; v8bf h[2]; } u;
        u.h[0] = *(const v8bf*)(bp);
        u.h[1] = *(const v8bf*)(bp + 16);
        return u.v;
    };

    v8f acc0 = {}, acc1 = {}, acc2 = {};

    // Prologue of the 2-stage pipeline.
    v16bf aC  = lda_frag(0);
    v16bf b0C = ldb_frag(0, n0 + ncol);
    v16bf b1C = ldb_frag(0, n0 + 16 + ncol);
    v16bf b2C = ldb_frag(0, n0 + 32 + ncol);

    for (int kk = 0; kk < KD_ / 32; ++kk) {
        v16bf aN = aC, b0N = b0C, b1N = b1C, b2N = b2C;
        if (kk + 1 < KD_ / 32) {
            aN  = lda_frag(kk + 1);
            b0N = ldb_frag(kk + 1, n0 + ncol);
            b1N = ldb_frag(kk + 1, n0 + 16 + ncol);
            b2N = ldb_frag(kk + 1, n0 + 32 + ncol);
        }
        acc0 = __builtin_amdgcn_wmma_f32_16x16x32_bf16(false, aC, false, b0C,
                                                       (short)0, acc0, false, false);
        acc1 = __builtin_amdgcn_wmma_f32_16x16x32_bf16(false, aC, false, b1C,
                                                       (short)0, acc1, false, false);
        acc2 = __builtin_amdgcn_wmma_f32_16x16x32_bf16(false, aC, false, b2C,
                                                       (short)0, acc2, false, false);
        aC = aN; b0C = b0N; b1C = b1N; b2C = b2N;
    }

    // Bias + ReLU -> LDS (C layout: VGPR r, lane -> M = r + 8*half, N = lane&15).
    #pragma unroll
    for (int r = 0; r < 8; ++r) {
        int mr = hl * 8 + r;
        float v;
        v = acc0[r] + bias[n0 + ncol];      ldsC[mr * F_ + n0 + ncol]      = fmaxf(v, 0.f);
        v = acc1[r] + bias[n0 + 16 + ncol]; ldsC[mr * F_ + n0 + 16 + ncol] = fmaxf(v, 0.f);
        v = acc2[r] + bias[n0 + 32 + ncol]; ldsC[mr * F_ + n0 + 32 + ncol] = fmaxf(v, 0.f);
    }
    __syncthreads();

    // LayerNorm over 384 features: each wave handles 2 rows.
    for (int rr = 0; rr < 2; ++rr) {
        int mr = wave * 2 + rr;
        float s = 0.f, s2 = 0.f;
        #pragma unroll
        for (int i = 0; i < 12; ++i) {
            float v = ldsC[mr * F_ + lane + 32 * i];
            s += v; s2 += v * v;
        }
        #pragma unroll
        for (int off = 16; off > 0; off >>= 1) {
            s  += __shfl_xor(s,  off, 32);
            s2 += __shfl_xor(s2, off, 32);
        }
        float mean = s * (1.f / F_);
        float var  = s2 * (1.f / F_) - mean * mean;
        float inv  = rsqrtf(var + 1e-5f);
        size_t base = ((size_t)b * T_ + t0 + mr) * F_;
        #pragma unroll
        for (int i = 0; i < 12; ++i) {
            int c = lane + 32 * i;
            float v = (ldsC[mr * F_ + c] - mean) * inv * gamma[c] + beta[c];
            Out[base + c] = (__bf16)v;
        }
    }
}

// ---------------------------------------------------------------------------
// Linear head: pred[token] = h[token,:] . wl + bl, masked. One wave per token.
// ---------------------------------------------------------------------------
__global__ __launch_bounds__(256)
void k_linear(const __bf16* __restrict__ h, const float* __restrict__ wl,
              const float* __restrict__ bl, const unsigned char* __restrict__ mask,
              float* __restrict__ out)
{
    int token = blockIdx.x * 8 + (threadIdx.x >> 5);
    int lane  = threadIdx.x & 31;
    const __bf16* row = h + (size_t)token * F_;
    float s = 0.f;
    #pragma unroll
    for (int i = 0; i < 12; ++i) {
        int c = lane + 32 * i;
        s += (float)row[c] * wl[c];
    }
    #pragma unroll
    for (int off = 16; off > 0; off >>= 1) s += __shfl_xor(s, off, 32);
    if (lane == 0) out[token] = mask[token] ? 0.f : (s + bl[0]);
}

// ---------------------------------------------------------------------------
// x[token,:] += emb[searchsorted_left(bins, pred[token]), :]. One wave/token.
// ---------------------------------------------------------------------------
__global__ __launch_bounds__(256)
void k_add_emb(float* __restrict__ x, const float* __restrict__ pred,
               const float* __restrict__ bins, const float* __restrict__ emb)
{
    int token = blockIdx.x * 8 + (threadIdx.x >> 5);
    int lane  = threadIdx.x & 31;
    float p = pred[token];
    int lo = 0, hi = 255;              // NBINS-1 boundaries
    while (lo < hi) {
        int mid = (lo + hi) >> 1;
        if (bins[mid] < p) lo = mid + 1; else hi = mid;
    }
    const float* e  = emb + (size_t)lo * H_;
    float*       xr = x + (size_t)token * H_;
    #pragma unroll
    for (int i = 0; i < 12; ++i) {
        int c = lane + 32 * i;
        xr[c] += e[c];
    }
}

// ---------------------------------------------------------------------------
// Per-batch cumsum of durations + per-frame searchsorted_right index table.
// ---------------------------------------------------------------------------
__global__ __launch_bounds__(256)
void k_cumidx(const int* __restrict__ dur, int* __restrict__ idxbuf,
              float* __restrict__ mel_out)
{
    __shared__ int cum[T_];
    int b = blockIdx.x, tid = threadIdx.x;
    for (int i = tid; i < T_; i += 256) cum[i] = dur[b * T_ + i];
    __syncthreads();
    if (tid == 0) {
        int s = 0;
        for (int i = 0; i < T_; ++i) { s += cum[i]; cum[i] = s; }
    }
    __syncthreads();
    int total = cum[T_ - 1];
    if (tid == 0) mel_out[b] = (float)total;
    for (int f = tid; f < MAXLEN_; f += 256) {
        int lo = 0, hi = T_;
        while (lo < hi) {
            int mid = (lo + hi) >> 1;
            if (cum[mid] <= f) lo = mid + 1; else hi = mid;
        }
        int idx = lo < T_ - 1 ? lo : T_ - 1;
        idxbuf[b * MAXLEN_ + f] = (f < total) ? idx : -1;
    }
}

// ---------------------------------------------------------------------------
// Expansion gather: one 96-thread block per output frame; one float4 per
// thread; NT stores (201 MB stream > 192 MB L2 -> don't rinse the cache).
// ---------------------------------------------------------------------------
__global__ __launch_bounds__(96)
void k_expand(const float* __restrict__ x, const int* __restrict__ idxbuf,
              float* __restrict__ out)
{
    int fb  = blockIdx.x;            // b*MAXLEN_ + frame
    int b   = fb >> 12;
    int idx = idxbuf[fb];
    v4f* op = (v4f*)(out + (size_t)fb * H_) + threadIdx.x;
    if (idx < 0) {
        v4f z = {0.f, 0.f, 0.f, 0.f};
        __builtin_nontemporal_store(z, op);
    } else {
        const v4f* ip = (const v4f*)(x + ((size_t)b * T_ + idx) * H_) + threadIdx.x;
        __builtin_nontemporal_store(*ip, op);
    }
}

// ---------------------------------------------------------------------------
extern "C" void kernel_launch(void* const* d_in, const int* in_sizes, int n_in,
                              void* d_out, int out_size, void* d_ws, size_t ws_size,
                              hipStream_t stream)
{
    (void)in_sizes; (void)n_in; (void)out_size; (void)ws_size;

    const float*         x        = (const float*)d_in[0];
    const int*           duration = (const int*)d_in[1];
    const unsigned char* src_mask = (const unsigned char*)d_in[2];
    // d_in[3] = max_len (constant 4096, unused)
    // Param dicts flattened: w1,b1,g1,be1,w2,b2,g2,be2,wl,bl per predictor.
    auto P = [&](int blk, int k) { return (const float*)d_in[4 + blk * 10 + k]; };
    const float* pitch_bins  = (const float*)d_in[34];
    const float* energy_bins = (const float*)d_in[35];
    const float* pitch_emb   = (const float*)d_in[36];
    const float* energy_emb  = (const float*)d_in[37];

    // Workspace carve-up (~56 MB).
    char* ws = (char*)d_ws;
    const size_t WSZ = (size_t)KD_ * F_;                  // 442368 elems per conv
    __bf16* Wt = (__bf16*)ws;
    size_t off = 6 * WSZ * sizeof(__bf16);
    __bf16* h1   = (__bf16*)(ws + off); off += (size_t)B_ * T_ * F_ * 2;
    __bf16* h2   = (__bf16*)(ws + off); off += (size_t)B_ * T_ * F_ * 2;
    float*  xcur = (float*)(ws + off);  off += (size_t)B_ * T_ * H_ * 4;
    int*    idxb = (int*)(ws + off);    off += (size_t)B_ * MAXLEN_ * 4;

    // Output layout: expanded | log_dur | pitch | energy | mel_len.
    float* out_exp    = (float*)d_out;
    float* out_ldur   = out_exp + (size_t)B_ * MAXLEN_ * H_;
    float* out_pitch  = out_ldur + B_ * T_;
    float* out_energy = out_pitch + B_ * T_;
    float* out_mel    = out_energy + B_ * T_;

    // 1) Convert the six conv weights to bf16, N-major.
    const float* wsrc[6] = { P(0,0), P(0,4), P(1,0), P(1,4), P(2,0), P(2,4) };
    for (int m = 0; m < 6; ++m)
        k_wt<<<(KD_ * F_ + 255) / 256, 256, 0, stream>>>(wsrc[m], Wt + (size_t)m * WSZ);

    // 2) Working copy of x (mutated by embedding adds).
    hipMemcpyAsync(xcur, x, (size_t)B_ * T_ * H_ * 4, hipMemcpyDeviceToDevice, stream);

    auto vp = [&](int blk, float* pred_out) {
        k_conv_ln<false><<<B_ * (T_ / 16), 256, 0, stream>>>(
            xcur, Wt + (size_t)(blk * 2 + 0) * WSZ, P(blk,1), P(blk,2), P(blk,3), h1);
        k_conv_ln<true><<<B_ * (T_ / 16), 256, 0, stream>>>(
            h1, Wt + (size_t)(blk * 2 + 1) * WSZ, P(blk,5), P(blk,6), P(blk,7), h2);
        k_linear<<<(B_ * T_) / 8, 256, 0, stream>>>(h2, P(blk,8), P(blk,9), src_mask, pred_out);
    };

    vp(0, out_ldur);                                              // duration
    vp(1, out_pitch);                                             // pitch
    k_add_emb<<<(B_ * T_) / 8, 256, 0, stream>>>(xcur, out_pitch, pitch_bins, pitch_emb);
    vp(2, out_energy);                                            // energy
    k_add_emb<<<(B_ * T_) / 8, 256, 0, stream>>>(xcur, out_energy, energy_bins, energy_emb);

    k_cumidx<<<B_, 256, 0, stream>>>(duration, idxb, out_mel);
    k_expand<<<B_ * MAXLEN_, 96, 0, stream>>>(xcur, idxb, out_exp);
}
